// EfficientRNN_13460427506295
// MI455X (gfx1250) — matrix operand order repaired and checked by reading, not compiled, not run on hardware
//
#include <hip/hip_runtime.h>
#include <hip/hip_bf16.h>
#include <math.h>

// Problem constants (from reference)
#define IN_  512
#define H_   512
#define L_   2
#define S_   3
#define B_   64
#define T_   512
#define G3H  1536            // 3*H  (gate rows r,z,n)
#define HST  (L_*H_)         // per-batch row stride of h state [B,L,H]

typedef __attribute__((ext_vector_type(16))) __bf16 v16bf;
typedef __attribute__((ext_vector_type(8)))  __bf16 bf16x8;
typedef __attribute__((ext_vector_type(8)))  float  v8f;
typedef __attribute__((ext_vector_type(8)))  float  f32x8;

// ---------------------------------------------------------------------------
// Fragment loaders (ISA 7.12.2 layouts, wave32)
// A 16x32 bf16: lanes 0-15 row=lane hold K = kt+[0..7], kt+[16..23];
//               lanes 16-31 row=lane-16 hold K = kt+[8..15], kt+[24..31].
__device__ __forceinline__ v16bf load_a_frag_f32(const float* __restrict__ row,
                                                 int kt, int khalf) {
  f32x8 lo = *(const f32x8*)(row + kt + khalf * 8);
  f32x8 hi = *(const f32x8*)(row + kt + 16 + khalf * 8);
  v16bf r;
#pragma unroll
  for (int e = 0; e < 8; ++e) {
    r[e]     = (__bf16)lo[e];
    r[8 + e] = (__bf16)hi[e];
  }
  return r;
}

__device__ __forceinline__ v16bf load_a_frag_bf16(const __bf16* __restrict__ row,
                                                  int kt, int khalf) {
  bf16x8 lo = *(const bf16x8*)(row + kt + khalf * 8);        // 16B aligned
  bf16x8 hi = *(const bf16x8*)(row + kt + 16 + khalf * 8);   // 16B aligned
  return __builtin_shufflevector(lo, hi, 0, 1, 2, 3, 4, 5, 6, 7,
                                 8, 9, 10, 11, 12, 13, 14, 15);
}

// B 32x16 bf16: lane n (0-15) holds col n, K = kt+[0..7], kt+[16..23];
//               lane n+16 holds col n, K = kt+[8..15], kt+[24..31].  Row-major weight row
//               == B column, so this is two 16B loads per lane.
__device__ __forceinline__ v16bf load_b_frag(const __bf16* __restrict__ wrow,
                                             int kt, int khalf) {
  bf16x8 lo = *(const bf16x8*)(wrow + kt + khalf * 8);
  bf16x8 hi = *(const bf16x8*)(wrow + kt + 16 + khalf * 8);
  return __builtin_shufflevector(lo, hi, 0, 1, 2, 3, 4, 5, 6, 7,
                                 8, 9, 10, 11, 12, 13, 14, 15);
}

__device__ __forceinline__ float sigmoid_f(float x) {
  return 1.0f / (1.0f + __expf(-x));
}

// ---------------------------------------------------------------------------
// One GRU layer for the selected expert.
// grid = 32 (16 output columns each), block = 32 (one wave).
// Each wave: 64 rows x 16 cols output; 6 gate GEMM tiles, K=512, bf16 WMMA.
// A operand is bf16 (ABF16=true) or f32 converted on the fly (fallback).
// Recurrent operand h_prev is always consumed as bf16; kernel emits both the
// f32 state (for router / elementwise carry) and a bf16 copy (next GEMM).
template <bool ABF16>
__global__ void __launch_bounds__(32)
gru_layer_kernel(const void* __restrict__ Av, long astride,      // GEMM input rows
                 const __bf16* __restrict__ hprev_bf,            // + b*HST + col
                 const float* __restrict__ hprev_f,              // + b*HST + col
                 float* __restrict__ hout_f,
                 __bf16* __restrict__ hout_bf,
                 const __bf16* __restrict__ WihBf,               // [S][L][3H][512]
                 const __bf16* __restrict__ WhhBf,               // [S][L][3H][512]
                 const float* __restrict__ bihAll,               // [S][L][3H]
                 const float* __restrict__ bhhAll,
                 const int* __restrict__ curPtr,
                 int layer,
                 float* __restrict__ out_t,                      // d_out + t*B*H (or null)
                 float* __restrict__ out_last) {                 // h_last slot (or null)
  const int lane  = threadIdx.x;
  const int jc    = blockIdx.x << 4;
  const int cur   = *curPtr;
  const size_t msz = (size_t)G3H * IN_;
  const __bf16* Wih = WihBf + (size_t)(cur * L_ + layer) * msz;
  const __bf16* Whh = WhhBf + (size_t)(cur * L_ + layer) * msz;
  const float*  bih = bihAll + (size_t)(cur * L_ + layer) * G3H;
  const float*  bhh = bhhAll + (size_t)(cur * L_ + layer) * G3H;

  const int col   = jc + (lane & 15);
  const int khalf = lane >> 4;
  const int rit   = lane & 15;

  // B operand row pointers (weight row g*512+col == B column for gate g)
  const __bf16* wir = Wih + (size_t)(col)        * IN_;
  const __bf16* wiz = Wih + (size_t)(col + 512)  * IN_;
  const __bf16* win = Wih + (size_t)(col + 1024) * IN_;
  const __bf16* whr = Whh + (size_t)(col)        * H_;
  const __bf16* whz = Whh + (size_t)(col + 512)  * H_;
  const __bf16* whn = Whh + (size_t)(col + 1024) * H_;

  v8f accR[4] = {}, accZ[4] = {}, accIN[4] = {}, accHN[4] = {};

  for (int kt = 0; kt < 512; kt += 32) {
    v16bf bir = load_b_frag(wir, kt, khalf);
    v16bf biz = load_b_frag(wiz, kt, khalf);
    v16bf bin = load_b_frag(win, kt, khalf);
    v16bf bhr = load_b_frag(whr, kt, khalf);
    v16bf bhz = load_b_frag(whz, kt, khalf);
    v16bf bhn = load_b_frag(whn, kt, khalf);
#pragma unroll
    for (int mt = 0; mt < 4; ++mt) {
      const int row = mt * 16 + rit;
      v16bf af;
      if constexpr (ABF16) {
        af = load_a_frag_bf16((const __bf16*)Av + (size_t)row * astride, kt, khalf);
      } else {
        af = load_a_frag_f32((const float*)Av + (size_t)row * astride, kt, khalf);
      }
      v16bf hf = load_a_frag_bf16(hprev_bf + (size_t)row * HST, kt, khalf);
      // r,z gates: gi + gh fused into one accumulator chain
      accR[mt] = __builtin_amdgcn_wmma_f32_16x16x32_bf16(false, af, false, bir,
                                                         (short)0, accR[mt], false, false);
      accR[mt] = __builtin_amdgcn_wmma_f32_16x16x32_bf16(false, hf, false, bhr,
                                                         (short)0, accR[mt], false, false);
      accZ[mt] = __builtin_amdgcn_wmma_f32_16x16x32_bf16(false, af, false, biz,
                                                         (short)0, accZ[mt], false, false);
      accZ[mt] = __builtin_amdgcn_wmma_f32_16x16x32_bf16(false, hf, false, bhz,
                                                         (short)0, accZ[mt], false, false);
      // n gate: gi_n and gh_n must stay separate (r multiplies gh_n)
      accIN[mt] = __builtin_amdgcn_wmma_f32_16x16x32_bf16(false, af, false, bin,
                                                          (short)0, accIN[mt], false, false);
      accHN[mt] = __builtin_amdgcn_wmma_f32_16x16x32_bf16(false, hf, false, bhn,
                                                          (short)0, accHN[mt], false, false);
    }
  }

  const float br  = bih[col]        + bhh[col];
  const float bz  = bih[col + 512]  + bhh[col + 512];
  const float bni = bih[col + 1024];
  const float bnh = bhh[col + 1024];

  // C/D layout: VGPR i -> row = i (+8 for lanes 16-31), col = lane&15
#pragma unroll
  for (int mt = 0; mt < 4; ++mt) {
#pragma unroll
    for (int i = 0; i < 8; ++i) {
      const int b   = mt * 16 + i + khalf * 8;
      const float r = sigmoid_f(accR[mt][i] + br);
      const float z = sigmoid_f(accZ[mt][i] + bz);
      const float n = tanhf(accIN[mt][i] + bni + r * (accHN[mt][i] + bnh));
      const float hp = hprev_f[(size_t)b * HST + col];
      const float h  = (1.0f - z) * n + z * hp;
      hout_f[(size_t)b * HST + col]  = h;
      hout_bf[(size_t)b * HST + col] = (__bf16)h;
      if (out_t)    out_t[(size_t)b * H_ + col]    = h;
      if (out_last) out_last[(size_t)b * H_ + col] = h;
    }
  }
}

// ---------------------------------------------------------------------------
// Router, algebraically reduced:
//   le[b,l]   = h[b,l,:]·wsum + bsum
//   shsum[h]  = sum_{b,l} le[b,l]*hstate[b,l,h];  xsum[i] = sum_b x_t[b,i]
//   logit[s]  = shsum·Wsel[s,:H] + xsum·Wsel[s,H:] + B*bsel[s]
//   probs = softmax(logit)*p ; cur = argmax ; p update with penalty 0.7
__global__ void __launch_bounds__(256)
router_kernel(const float* __restrict__ hstate,   // [B*L, H]
              const float* __restrict__ x_t,      // + b*(T*IN) rows
              const float* __restrict__ wsum,
              const float* __restrict__ bsumPtr,
              const float* __restrict__ Wsel,     // [S, H+IN]
              const float* __restrict__ bsel,
              float* __restrict__ p,
              int* __restrict__ curPtr) {
  __shared__ float le[B_ * L_];
  __shared__ float vsum[H_ + IN_];
  __shared__ float red[256];
  __shared__ float logits[S_];
  const int tid = threadIdx.x;

  if (tid < B_ * L_) {
    const float* hrow = hstate + (size_t)tid * H_;
    float acc = 0.f;
    for (int k = 0; k < H_; ++k) acc += hrow[k] * wsum[k];
    le[tid] = acc + *bsumPtr;
  }
  __syncthreads();

  for (int hid = tid; hid < H_; hid += 256) {
    float acc = 0.f;
    for (int q = 0; q < B_ * L_; ++q) acc += le[q] * hstate[(size_t)q * H_ + hid];
    vsum[hid] = acc;
  }
  for (int i = tid; i < IN_; i += 256) {
    float acc = 0.f;
    for (int b = 0; b < B_; ++b) acc += x_t[(size_t)b * (T_ * IN_) + i];
    vsum[H_ + i] = acc;
  }
  __syncthreads();

  for (int s = 0; s < S_; ++s) {
    float acc = 0.f;
    for (int k = tid; k < H_ + IN_; k += 256) acc += vsum[k] * Wsel[s * (H_ + IN_) + k];
    red[tid] = acc;
    __syncthreads();
    for (int off = 128; off > 0; off >>= 1) {
      if (tid < off) red[tid] += red[tid + off];
      __syncthreads();
    }
    if (tid == 0) logits[s] = red[0] + (float)B_ * bsel[s];
    __syncthreads();
  }

  if (tid == 0) {
    float m = fmaxf(logits[0], fmaxf(logits[1], logits[2]));
    float e0 = __expf(logits[0] - m), e1 = __expf(logits[1] - m), e2 = __expf(logits[2] - m);
    float inv = 1.0f / (e0 + e1 + e2);
    float pr0 = e0 * inv * p[0], pr1 = e1 * inv * p[1], pr2 = e2 * inv * p[2];
    int cur = 0; float best = pr0;                      // jnp.argmax: first max wins
    if (pr1 > best) { best = pr1; cur = 1; }
    if (pr2 > best) { best = pr2; cur = 2; }
    *curPtr = cur;
    float n0 = p[0] * (cur == 0 ? 0.7f : 1.0f);
    float n1 = p[1] * (cur == 1 ? 0.7f : 1.0f);
    float n2 = p[2] * (cur == 2 ? 0.7f : 1.0f);
    float mx = fmaxf(n0, fmaxf(n1, n2));
    p[0] = n0 / mx; p[1] = n1 / mx; p[2] = n2 / mx;
  }
}

// ---------------------------------------------------------------------------
__global__ void __launch_bounds__(256)
init_kernel(float* __restrict__ hA, __bf16* __restrict__ hAbf,
            const float* __restrict__ Wlw, const float* __restrict__ blw,
            float* __restrict__ wsum, float* __restrict__ bsum,
            float* __restrict__ p, int* __restrict__ curPtr) {
  const int tid = blockIdx.x * blockDim.x + threadIdx.x;
  for (int i = tid; i < B_ * L_ * H_; i += gridDim.x * blockDim.x) {
    hA[i]   = 0.0f;
    hAbf[i] = (__bf16)0.0f;
  }
  if (tid < H_) {
    float a = 0.f;
    for (int g = 0; g < H_; ++g) a += Wlw[(size_t)g * H_ + tid];
    wsum[tid] = a;
  }
  if (tid == 0) {
    float a = 0.f;
    for (int g = 0; g < H_; ++g) a += blw[g];
    *bsum = a;
    p[0] = 0.7f; p[1] = 1.0f; p[2] = 1.0f;   // post-step-0 penalty state
    *curPtr = 0;                              // expert 0 at t=0
  }
}

__global__ void __launch_bounds__(256)
convert_weights_kernel(const float* __restrict__ Wih_first,   // [S,3H,IN]
                       const float* __restrict__ Wih_rest,    // [S,L-1,3H,H]
                       const float* __restrict__ Whh,         // [S,L,3H,H]
                       __bf16* __restrict__ WihBf,            // [S,L,3H,512]
                       __bf16* __restrict__ WhhBf) {
  const size_t msz = (size_t)G3H * IN_;
  const size_t N   = (size_t)S_ * L_ * msz;
  for (size_t idx = (size_t)blockIdx.x * blockDim.x + threadIdx.x; idx < N;
       idx += (size_t)gridDim.x * blockDim.x) {
    size_t s   = idx / (L_ * msz);
    size_t rem = idx % (L_ * msz);
    size_t l   = rem / msz;
    size_t o   = rem % msz;
    float vih = (l == 0) ? Wih_first[s * msz + o]
                         : Wih_rest[(s * (L_ - 1) + (l - 1)) * msz + o];
    WihBf[idx] = (__bf16)vih;
    WhhBf[idx] = (__bf16)Whh[idx];
  }
}

__global__ void __launch_bounds__(256)
convert_x_kernel(const float* __restrict__ x, __bf16* __restrict__ xbf) {
  const size_t N = (size_t)B_ * T_ * IN_;
  for (size_t i = (size_t)blockIdx.x * blockDim.x + threadIdx.x; i < N;
       i += (size_t)gridDim.x * blockDim.x) {
    xbf[i] = (__bf16)x[i];
  }
}

// ---------------------------------------------------------------------------
extern "C" void kernel_launch(void* const* d_in, const int* in_sizes, int n_in,
                              void* d_out, int out_size, void* d_ws, size_t ws_size,
                              hipStream_t stream) {
  const float* x         = (const float*)d_in[0];
  const float* Wih_first = (const float*)d_in[1];
  const float* Wih_rest  = (const float*)d_in[2];
  const float* Whh       = (const float*)d_in[3];
  const float* bih       = (const float*)d_in[4];
  const float* bhh       = (const float*)d_in[5];
  const float* Wlw       = (const float*)d_in[6];
  const float* blw       = (const float*)d_in[7];
  const float* Wsel      = (const float*)d_in[8];
  const float* bsel      = (const float*)d_in[9];
  float* out = (float*)d_out;

  // Workspace layout (all 32B-aligned); xBf is last so we can fall back if
  // the workspace is too small for the 32MB bf16 copy of x.
  const size_t wbytes  = (size_t)S_ * L_ * G3H * IN_ * sizeof(__bf16);  // 9,437,184 B
  const size_t hfB     = (size_t)B_ * L_ * H_ * sizeof(float);          // 262,144 B
  const size_t hbfB    = (size_t)B_ * L_ * H_ * sizeof(__bf16);         // 131,072 B
  const size_t xbfB    = (size_t)B_ * T_ * IN_ * sizeof(__bf16);        // 33,554,432 B

  char* base = (char*)d_ws;
  size_t off = 0;
  __bf16* WihBf = (__bf16*)(base + off); off += wbytes;
  __bf16* WhhBf = (__bf16*)(base + off); off += wbytes;
  float*  hA    = (float*)(base + off);  off += hfB;
  float*  hB    = (float*)(base + off);  off += hfB;
  __bf16* hAbf  = (__bf16*)(base + off); off += hbfB;
  __bf16* hBbf  = (__bf16*)(base + off); off += hbfB;
  float*  wsum  = (float*)(base + off);  off += H_ * sizeof(float);
  float*  bsum  = (float*)(base + off);  off += 64;   // keep 32B alignment
  float*  p     = bsum + 1;
  int*    cur   = (int*)(p + S_);
  __bf16* xBf   = (__bf16*)(base + off);
  const bool use_xbf = (ws_size >= off + xbfB);

  init_kernel<<<64, 256, 0, stream>>>(hA, hAbf, Wlw, blw, wsum, bsum, p, cur);
  convert_weights_kernel<<<4096, 256, 0, stream>>>(Wih_first, Wih_rest, Whh, WihBf, WhhBf);
  if (use_xbf) convert_x_kernel<<<8192, 256, 0, stream>>>(x, xBf);

  for (int t = 0; t < T_; ++t) {
    const float*  hRead   = (t & 1) ? hB : hA;
    float*        hWrite  = (t & 1) ? hA : hB;
    const __bf16* hReadBf = (t & 1) ? hBbf : hAbf;
    __bf16*       hWriteBf= (t & 1) ? hAbf : hBbf;

    if (t > 0) {
      router_kernel<<<1, 256, 0, stream>>>(hRead, x + (size_t)t * IN_,
                                           wsum, bsum, Wsel, bsel, p, cur);
    }
    // layer 0: A = x_t (row stride T*IN), hidden slice l=0
    if (use_xbf) {
      gru_layer_kernel<true><<<32, 32, 0, stream>>>(
          (const void*)(xBf + (size_t)t * IN_), (long)(T_ * IN_),
          hReadBf + 0 * H_, hRead + 0 * H_,
          hWrite + 0 * H_, hWriteBf + 0 * H_,
          WihBf, WhhBf, bih, bhh, cur, 0, nullptr, nullptr);
    } else {
      gru_layer_kernel<false><<<32, 32, 0, stream>>>(
          (const void*)(x + (size_t)t * IN_), (long)(T_ * IN_),
          hReadBf + 0 * H_, hRead + 0 * H_,
          hWrite + 0 * H_, hWriteBf + 0 * H_,
          WihBf, WhhBf, bih, bhh, cur, 0, nullptr, nullptr);
    }
    // layer 1: A = freshly written layer-0 h in bf16 (row stride L*H)
    const bool last = (t == T_ - 1);
    gru_layer_kernel<true><<<32, 32, 0, stream>>>(
        (const void*)(hWriteBf + 0 * H_), (long)HST,
        hReadBf + 1 * H_, hRead + 1 * H_,
        hWrite + 1 * H_, hWriteBf + 1 * H_,
        WihBf, WhhBf, bih, bhh, cur, 1,
        out + (size_t)t * B_ * H_,
        last ? out + (size_t)T_ * B_ * H_ : nullptr);
  }
}
